// FasterRCNN_64518998720523
// MI455X (gfx1250) — matrix operand order repaired and checked by reading, not compiled, not run on hardware
//
#include <hip/hip_runtime.h>
#include <hip/hip_bf16.h>
#include <math.h>

typedef __attribute__((ext_vector_type(16))) __bf16 v16bf;
typedef __attribute__((ext_vector_type(8)))  float  v8f;
typedef __attribute__((ext_vector_type(4)))  unsigned int u32x4;
typedef __attribute__((ext_vector_type(8)))  int i32x8;
typedef __attribute__((ext_vector_type(4)))  int i32x4;

#define BATCH 2
#define CIN   512
#define HGT   25
#define WID   25
#define NPOS  (HGT*WID)      /* 625  */
#define COUT  512
#define KDIM  (CIN*9)        /* 4608 */
#define NANC  9
#define AALL  (NPOS*NANC)    /* 5625 */
#define NHEAD 48             /* 36 reg + 9 cls + 3 pad */
#define SORTN 8192
#define CMT   64             /* conv block M tile (positions) */
#define CNT   128            /* conv block N tile (out channels) */

// ---------------- WMMA fragment loaders (contiguous k = 16*(lane/16) + j, same for A and B) ----------------
// A 16x32 bf16: lane holds row m=lane&15; elems j -> K=khi+j, khi=(lane>=16)?16:0.
__device__ inline v16bf frag_a(const __bf16* row, int lane) {
  int khi = (lane & 16) ? 16 : 0;
  v16bf a;
#pragma unroll
  for (int j = 0; j < 16; ++j) { a[j] = row[khi + j]; }
  return a;
}
// B 32x16: lane holds col n=lane&15; elems j -> K=kb+j, kb=(lane>=16)?16:0.
__device__ inline v16bf frag_b(const __bf16* row) {
  v16bf b;
#pragma unroll
  for (int j = 0; j < 16; ++j) b[j] = row[j];
  return b;
}
// f32 source (TDM-staged fp32 tile), convert on read
__device__ inline v16bf frag_b_f32(const float* row) {
  v16bf b;
#pragma unroll
  for (int j = 0; j < 16; ++j) b[j] = (__bf16)row[j];
  return b;
}

// ---------------- Kernel 1: 3x3 conv as implicit GEMM (bf16 WMMA + TDM) -----
// Block: 256 thr (8 waves). Tile: 64 positions x 128 out-channels.
// wave w: M sub-tile = (w>>1)*16, N half = (w&1)*64 -> 4 accumulators.
// Weight tiles (128 rows x 32 k, fp32) are streamed by the Tensor Data Mover
// into a ping-pong LDS buffer, one tile ahead of the matrix pipe.
__global__ __launch_bounds__(256)
void conv_wmma(const float* __restrict__ x, const float* __restrict__ w,
               const float* __restrict__ bias, __bf16* __restrict__ h) {
  __shared__ __bf16 Asm[CMT * 32];      // 4 KB  (im2col activations, bf16)
  __shared__ float  Bsm[2][CNT * 32];   // 32 KB (weights, fp32 via TDM, 2-deep)
  const int nblk = blockIdx.x;          // 0..3
  const int mblk = blockIdx.y;          // 0..9
  const int b    = blockIdx.z;
  const int tid  = threadIdx.x;
  const int lane = tid & 31, wave = tid >> 5;
  const int n0 = nblk * CNT;
  const int m0 = mblk * CMT;
  const int mt = (wave >> 1) * 16;      // M sub-tile offset within block
  const int nh = (wave & 1) * 64;       // N half offset within block

  // TDM D# group1: 2D tensor 512 x 4608 fp32, tile 128 rows x 32 elems.
  i32x8 g1;
  g1[0] = (2 << 16);                    // data_size=2 (4B)
  g1[1] = (KDIM & 0xFFFF) << 16;        // tensor_dim0[15:0] (4608)
  g1[2] = (COUT << 16) | (KDIM >> 16);  // tensor_dim1[15:0] | tensor_dim0[31:16]
  g1[3] = (32 << 16);                   // tile_dim0=32
  g1[4] = CNT;                          // tile_dim1=128, tile_dim2=0
  g1[5] = KDIM;                         // tensor_dim0_stride[31:0]
  g1[6] = 0;
  g1[7] = 0;
  const i32x4 gz4 = {0, 0, 0, 0};
  const i32x8 gz8 = {0, 0, 0, 0, 0, 0, 0, 0};
  const unsigned ldsB[2] = {(unsigned)(uintptr_t)(void*)&Bsm[0][0],
                            (unsigned)(uintptr_t)(void*)&Bsm[1][0]};

  auto issue_tdm = [&](int ks, unsigned lds) {
    unsigned long long ga = (unsigned long long)w
                          + ((unsigned long long)n0 * KDIM + ks) * 4ull;
    u32x4 g0;
    g0[0] = 1u;                                  // count=1, user descriptor
    g0[1] = lds;                                 // lds_addr
    g0[2] = (unsigned)ga;                        // global_addr[31:0]
    g0[3] = (unsigned)(ga >> 32) | 0x80000000u;  // addr[56:32] | type=2<<30
    __builtin_amdgcn_tensor_load_to_lds(g0, g1, gz4, gz4, gz8, 0);
  };

  const int T = KDIM / 32;              // 144 K-tiles
  if (wave == 0) issue_tdm(0, ldsB[0]);

  v8f acc0 = {}, acc1 = {}, acc2 = {}, acc3 = {};
  for (int t = 0; t < T; ++t) {
    const int ks = t * 32;
    // ---- prefetch next weight tile via TDM (overlaps with everything below)
    if (wave == 0 && t + 1 < T) issue_tdm(ks + 32, ldsB[(t + 1) & 1]);
    // ---- A tile (im2col gather): 64 positions x 32 k, bf16
    for (int i = tid; i < CMT * 32; i += 256) {
      int m = i >> 5, kk = i & 31;
      int p = m0 + m;
      float v = 0.f;
      if (p < NPOS) {
        int k = ks + kk;
        int c = k / 9, r = k - c * 9;
        int y  = p / WID + r / 3 - 1;
        int xx = p % WID + r % 3 - 1;
        if ((unsigned)y < (unsigned)HGT && (unsigned)xx < (unsigned)WID)
          v = x[((b * CIN + c) * HGT + y) * WID + xx];
      }
      Asm[m * 32 + kk] = (__bf16)v;
    }
    // ---- wait for tile t (in-order TENSORcnt: <=1 outstanding => t landed)
    if (wave == 0) {
      if (t + 1 < T) __builtin_amdgcn_s_wait_tensorcnt(1);
      else           __builtin_amdgcn_s_wait_tensorcnt(0);
    }
    __syncthreads();
    // ---- 4 WMMAs per wave on the staged tiles
    const float* Bcur = Bsm[t & 1];
    v16bf af = frag_a(Asm + (mt + (lane & 15)) * 32, lane);
    const int kb = (lane & 16) ? 16 : 0;
    v16bf b0 = frag_b_f32(Bcur + (nh +  0 + (lane & 15)) * 32 + kb);
    v16bf b1 = frag_b_f32(Bcur + (nh + 16 + (lane & 15)) * 32 + kb);
    v16bf b2 = frag_b_f32(Bcur + (nh + 32 + (lane & 15)) * 32 + kb);
    v16bf b3 = frag_b_f32(Bcur + (nh + 48 + (lane & 15)) * 32 + kb);
    acc0 = __builtin_amdgcn_wmma_f32_16x16x32_bf16(false, af, false, b0, (short)0, acc0, false, false);
    acc1 = __builtin_amdgcn_wmma_f32_16x16x32_bf16(false, af, false, b1, (short)0, acc1, false, false);
    acc2 = __builtin_amdgcn_wmma_f32_16x16x32_bf16(false, af, false, b2, (short)0, acc2, false, false);
    acc3 = __builtin_amdgcn_wmma_f32_16x16x32_bf16(false, af, false, b3, (short)0, acc3, false, false);
    __syncthreads();   // compute done: safe to overwrite A and B[(t+2)&1]==B[t&1]
  }
  // D layout: n = lane&15 (+tile), row m = r + (lane>=16 ? 8 : 0); h is bf16
  const int mb = (lane & 16) ? 8 : 0;
  const int nn = lane & 15;
#pragma unroll
  for (int r = 0; r < 8; ++r) {
    int p = m0 + mt + mb + r;
    if (p < NPOS) {
      __bf16* dst = h + ((size_t)b * NPOS + p) * COUT + n0 + nh;
      float v0 = acc0[r] + bias[n0 + nh +  0 + nn];
      float v1 = acc1[r] + bias[n0 + nh + 16 + nn];
      float v2 = acc2[r] + bias[n0 + nh + 32 + nn];
      float v3 = acc3[r] + bias[n0 + nh + 48 + nn];
      dst[ 0 + nn] = (__bf16)(v0 > 0.f ? v0 : 0.f);
      dst[16 + nn] = (__bf16)(v1 > 0.f ? v1 : 0.f);
      dst[32 + nn] = (__bf16)(v2 > 0.f ? v2 : 0.f);
      dst[48 + nn] = (__bf16)(v3 > 0.f ? v3 : 0.f);
    }
  }
}

// ---------------- Kernel 2: merged reg+cls head GEMM (bf16 WMMA + TDM A) ----
// h is bf16 [b][p][512]; activation tile (128 rows x 32 k, 2B) is a regular 2D
// tile -> TDM it straight into LDS. Rows past p=624 stage garbage but each A
// row only feeds its own output row, which is store-guarded.
__global__ __launch_bounds__(256)
void head_wmma(const __bf16* __restrict__ h, const float* __restrict__ reg_w,
               const float* __restrict__ reg_b, const float* __restrict__ cls_w,
               const float* __restrict__ cls_b, float* __restrict__ head) {
  __shared__ __bf16 Wc[NHEAD * 512];     // 48 KB
  __shared__ __bf16 Atile[128 * 32];     // 8 KB (TDM target)
  __shared__ float  biasS[NHEAD];
  const int bm = blockIdx.x;             // 0..4 (128 positions each)
  const int b  = blockIdx.y;
  const int tid = threadIdx.x, lane = tid & 31, wave = tid >> 5;
  for (int i = tid; i < NHEAD * 512; i += 256) {
    int n = i >> 9, k = i & 511;
    float v = 0.f;
    if (n < 36)      v = reg_w[n * 512 + k];
    else if (n < 45) v = cls_w[(n - 36) * 512 + k];
    Wc[i] = (__bf16)v;
  }
  if (tid < NHEAD)
    biasS[tid] = (tid < 36) ? reg_b[tid] : ((tid < 45) ? cls_b[tid - 36] : 0.f);
  const int m0 = bm * 128;

  i32x8 g1;
  g1[0] = (1 << 16);                    // data_size=1 (2B)
  g1[1] = (COUT & 0xFFFF) << 16;        // tensor_dim0 = 512
  g1[2] = (128 << 16);                  // tensor_dim1 = 128 (tile rows)
  g1[3] = (32 << 16);                   // tile_dim0 = 32
  g1[4] = 128;                          // tile_dim1 = 128
  g1[5] = COUT;                         // tensor_dim0_stride = 512
  g1[6] = 0;
  g1[7] = 0;
  const i32x4 gz4 = {0, 0, 0, 0};
  const i32x8 gz8 = {0, 0, 0, 0, 0, 0, 0, 0};
  const unsigned ldsA = (unsigned)(uintptr_t)(void*)Atile;

  v8f a0 = {}, a1 = {}, a2 = {};
  for (int ks = 0; ks < 512; ks += 32) {
    __syncthreads();   // Wc ready (first iter) / Atile consumed (later iters)
    if (wave == 0) {
      unsigned long long ga = (unsigned long long)h
                            + (((unsigned long long)b * NPOS + m0) * COUT + ks) * 2ull;
      u32x4 g0;
      g0[0] = 1u;
      g0[1] = ldsA;
      g0[2] = (unsigned)ga;
      g0[3] = (unsigned)(ga >> 32) | 0x80000000u;
      __builtin_amdgcn_tensor_load_to_lds(g0, g1, gz4, gz4, gz8, 0);
      __builtin_amdgcn_s_wait_tensorcnt(0);
    }
    __syncthreads();
    v16bf af = frag_a(Atile + (wave * 16 + (lane & 15)) * 32, lane);
    int kb = (lane & 16) ? 16 : 0;
    v16bf b0 = frag_b(Wc + ( 0 + (lane & 15)) * 512 + ks + kb);
    v16bf b1 = frag_b(Wc + (16 + (lane & 15)) * 512 + ks + kb);
    v16bf b2 = frag_b(Wc + (32 + (lane & 15)) * 512 + ks + kb);
    a0 = __builtin_amdgcn_wmma_f32_16x16x32_bf16(false, af, false, b0, (short)0, a0, false, false);
    a1 = __builtin_amdgcn_wmma_f32_16x16x32_bf16(false, af, false, b1, (short)0, a1, false, false);
    a2 = __builtin_amdgcn_wmma_f32_16x16x32_bf16(false, af, false, b2, (short)0, a2, false, false);
  }
  const int mb = (lane & 16) ? 8 : 0;
  const int nn = lane & 15;
#pragma unroll
  for (int r = 0; r < 8; ++r) {
    int p = m0 + wave * 16 + mb + r;
    if (p < NPOS) {
      float* dst = head + ((size_t)b * NPOS + p) * NHEAD;
      dst[nn]      = a0[r] + biasS[nn];
      dst[16 + nn] = a1[r] + biasS[16 + nn];
      dst[32 + nn] = a2[r] + biasS[32 + nn];
    }
  }
}

// ---------------- Kernel 3: anchor decode + sigmoid -------------------------
__global__ void decode_kernel(const float* __restrict__ head,
                              const float* __restrict__ anchors,
                              float* __restrict__ boxes, float* __restrict__ probs,
                              int* __restrict__ valid) {
  int idx = blockIdx.x * blockDim.x + threadIdx.x;
  if (idx >= BATCH * AALL) return;
  int b = idx / AALL, a = idx - b * AALL;
  int p = a / NANC, na = a - p * NANC;
  const float* hd = head + ((size_t)b * NPOS + p) * NHEAD;
  float ox = hd[na * 4 + 0], oy = hd[na * 4 + 1];
  float ow = hd[na * 4 + 2], oh = hd[na * 4 + 3];
  float logit = hd[36 + na];
  float prob = 1.f / (1.f + __expf(-logit));
  const float* an = anchors + (size_t)a * 4;
  float acx = (an[0] + an[2]) * 0.5f, acy = (an[1] + an[3]) * 0.5f;
  float aw = an[2] - an[0], ah = an[3] - an[1];
  float cx = ox * aw * 0.1f + acx;
  float cy = oy * ah * 0.1f + acy;
  float bw = __expf(ow * 0.2f) * aw;
  float bh = __expf(oh * 0.2f) * ah;
  float* bo = boxes + (size_t)idx * 4;
  bo[0] = cx - bw * 0.5f; bo[1] = cy - bh * 0.5f;
  bo[2] = cx + bw * 0.5f; bo[3] = cy + bh * 0.5f;
  probs[idx] = prob;
  valid[idx] = prob > 0.5f ? 1 : 0;
}

// ---------------- Kernel 4: per-image bitonic argsort (LDS) -----------------
__global__ __launch_bounds__(1024)
void sort_kernel(const float* __restrict__ probs, const int* __restrict__ valid,
                 int* __restrict__ order) {
  __shared__ float key[SORTN];
  __shared__ int   idxs[SORTN];
  const int b = blockIdx.x, tid = threadIdx.x;
  for (int i = tid; i < SORTN; i += 1024) {
    float k;
    if (i < AALL) {
      float pr = probs[b * AALL + i];
      k = valid[b * AALL + i] ? pr : -1.f;
      k = -k;
    } else k = 1e30f;
    key[i] = k; idxs[i] = i;
  }
  __syncthreads();
  for (int kk = 2; kk <= SORTN; kk <<= 1) {
    for (int j = kk >> 1; j > 0; j >>= 1) {
      for (int i = tid; i < SORTN; i += 1024) {
        int ixj = i ^ j;
        if (ixj > i) {
          bool up = ((i & kk) == 0);
          float ka = key[i], kb = key[ixj];
          int ia = idxs[i], ib = idxs[ixj];
          bool gt = (ka > kb) || (ka == kb && ia > ib);
          if (up == gt) { key[i] = kb; key[ixj] = ka; idxs[i] = ib; idxs[ixj] = ia; }
        }
      }
      __syncthreads();
    }
  }
  for (int i = tid; i < AALL; i += 1024) order[b * AALL + i] = idxs[i];
}

// ---------------- Kernel 5: gather + greedy NMS + emit output ---------------
__global__ __launch_bounds__(1024)
void nms_kernel(const float* __restrict__ boxes, const float* __restrict__ probs,
                const int* __restrict__ valid, const int* __restrict__ order,
                float* __restrict__ sb, float* __restrict__ sarea,
                float* __restrict__ out) {
  __shared__ int sup[AALL];            // 22.5 KB
  const int b = blockIdx.x, tid = threadIdx.x;
  const int base = b * AALL;
  float* sbb = sb + (size_t)base * 4;
  float* sar = sarea + base;
  for (int i = tid; i < AALL; i += 1024) {
    int o = order[base + i];
    const float* src = boxes + ((size_t)base + o) * 4;
    float x1 = src[0], y1 = src[1], x2 = src[2], y2 = src[3];
    sbb[i * 4 + 0] = x1; sbb[i * 4 + 1] = y1;
    sbb[i * 4 + 2] = x2; sbb[i * 4 + 3] = y2;
    sar[i] = (x2 - x1) * (y2 - y1);
    sup[i] = valid[base + o] ? 0 : 1;
    float* od = out + ((size_t)base + i) * 6;
    od[0] = x1; od[1] = y1; od[2] = x2; od[3] = y2; od[4] = probs[base + o];
  }
  __threadfence_block();
  __syncthreads();
  for (int i = 0; i < AALL; ++i) {
    if (sup[i] == 0) {                 // uniform across block
      float x1 = sbb[i * 4 + 0], y1 = sbb[i * 4 + 1];
      float x2 = sbb[i * 4 + 2], y2 = sbb[i * 4 + 3];
      float ai = sar[i];
      for (int j = i + 1 + tid; j < AALL; j += 1024) {
        if (sup[j] == 0) {
          float iw = fminf(x2, sbb[j * 4 + 2]) - fmaxf(x1, sbb[j * 4 + 0]);
          float ih = fminf(y2, sbb[j * 4 + 3]) - fmaxf(y1, sbb[j * 4 + 1]);
          iw = iw > 0.f ? iw : 0.f; ih = ih > 0.f ? ih : 0.f;
          float inter = iw * ih;
          float ov = inter / (ai + sar[j] - inter);
          if (ov > 0.5f) sup[j] = 1;
        }
      }
    }
    __syncthreads();
  }
  for (int i = tid; i < AALL; i += 1024)
    out[((size_t)base + i) * 6 + 5] = (sup[i] == 0) ? 1.f : 0.f;
}

// ---------------- launch --------------------------------------------------
extern "C" void kernel_launch(void* const* d_in, const int* in_sizes, int n_in,
                              void* d_out, int out_size, void* d_ws, size_t ws_size,
                              hipStream_t stream) {
  const float* x      = (const float*)d_in[0];
  const float* conv_w = (const float*)d_in[1];
  const float* conv_b = (const float*)d_in[2];
  const float* reg_w  = (const float*)d_in[3];
  const float* reg_b  = (const float*)d_in[4];
  const float* cls_w  = (const float*)d_in[5];
  const float* cls_b  = (const float*)d_in[6];
  const float* anchors= (const float*)d_in[7];

  __bf16* h     = (__bf16*)d_ws;                      // B*625*512 (bf16)
  float* head   = (float*)(h + (size_t)BATCH * NPOS * COUT);
  float* boxes  = head   + (size_t)BATCH * NPOS * NHEAD;
  float* probs  = boxes  + (size_t)BATCH * AALL * 4;
  float* sboxes = probs  + (size_t)BATCH * AALL;
  float* sarea  = sboxes + (size_t)BATCH * AALL * 4;
  int*   valid  = (int*)(sarea + (size_t)BATCH * AALL);
  int*   order  = valid  + (size_t)BATCH * AALL;

  conv_wmma<<<dim3(COUT / CNT, (NPOS + CMT - 1) / CMT, BATCH), 256, 0, stream>>>(
      x, conv_w, conv_b, h);
  head_wmma<<<dim3((NPOS + 127) / 128, BATCH), 256, 0, stream>>>(
      h, reg_w, reg_b, cls_w, cls_b, head);
  decode_kernel<<<(BATCH * AALL + 255) / 256, 256, 0, stream>>>(
      head, anchors, boxes, probs, valid);
  sort_kernel<<<BATCH, 1024, 0, stream>>>(probs, valid, order);
  nms_kernel<<<BATCH, 1024, 0, stream>>>(boxes, probs, valid, order,
                                         sboxes, sarea, (float*)d_out);
}
